// SlotAttentionAnchor_18854906429598
// MI455X (gfx1250) — matrix operand order repaired and checked by reading, not compiled, not run on hardware
//
#include <hip/hip_runtime.h>
#include <math.h>

typedef _Float16 h16;
typedef __attribute__((ext_vector_type(16))) _Float16 v16h;
typedef __attribute__((ext_vector_type(8)))  _Float16 v8h;
typedef __attribute__((ext_vector_type(8)))  float    v8f;

constexpr int   BN     = 8;
constexpr int   NPIX   = 4096;   // 64*64
constexpr int   KFG    = 8;      // foreground slots
constexpr int   NSL    = 9;      // 1 bg + 8 fg
constexpr int   NITER  = 4;
constexpr float SCALEF = 0.125f; // 64^-0.5
constexpr float MOMF   = 0.5f;
constexpr float EPSF   = 1e-8f;
constexpr float LNEPS  = 1e-5f;

constexpr size_t OUT_ATTN = 5184;    // after slots (8*9*72)
constexpr size_t OUT_POS  = 300096;  // after attn  (8*9*4096)

// ---- workspace arena (all offsets multiples of 256 bytes) ----
constexpr size_t OFF_AH   = 0;                                          // 4 folded 64x64 mats, f16
constexpr size_t OFF_WGMF = OFF_AH   + (size_t)4*64*64*sizeof(h16);
constexpr size_t OFF_WGMB = OFF_WGMF + (size_t)64*4*sizeof(float);
constexpr size_t OFF_BGMF = OFF_WGMB + (size_t)64*4*sizeof(float);
constexpr size_t OFF_BGMB = OFF_BGMF + (size_t)64*sizeof(float);
constexpr size_t OFF_KKH  = OFF_BGMB + (size_t)64*sizeof(float);        // [b][n][d] f16
constexpr size_t OFF_KBGH = OFF_KKH  + (size_t)BN*NPIX*64*sizeof(h16);  // [b][n][d] f16
constexpr size_t OFF_VVT  = OFF_KBGH + (size_t)BN*NPIX*64*sizeof(h16);  // [b][d][n] f16
constexpr size_t OFF_VBGT = OFF_VVT  + (size_t)BN*NPIX*64*sizeof(h16);  // [b][d][n] f16
constexpr size_t OFF_FCT  = OFF_VBGT + (size_t)BN*NPIX*64*sizeof(h16);  // [b][16][n] f16 misc operand
constexpr size_t OFF_QH   = OFF_FCT  + (size_t)BN*16*NPIX*sizeof(h16);  // [b][16][64] f16
constexpr size_t OFF_PM   = OFF_QH   + (size_t)BN*16*64*sizeof(h16);    // [b][8][64] f32
constexpr size_t OFF_CB   = OFF_PM   + (size_t)BN*8*64*sizeof(float);   // [b][16] f32
constexpr size_t OFF_SB   = OFF_CB   + (size_t)BN*16*sizeof(float);     // slots [b][16][64] f32
constexpr size_t OFF_PPOS = OFF_SB   + (size_t)BN*16*64*sizeof(float);  // [b][8][2] f32
constexpr size_t OFF_ATTN = OFF_PPOS + (size_t)BN*8*2*sizeof(float);    // [b][16][n] f16
constexpr size_t OFF_SSUM = OFF_ATTN + (size_t)BN*16*NPIX*sizeof(h16);  // [b][16] f32
constexpr size_t OFF_SAFG = OFF_SSUM + (size_t)BN*16*sizeof(float);     // [b][16][64] f32
constexpr size_t OFF_SABG = OFF_SAFG + (size_t)BN*16*64*sizeof(float);  // [b][16][64] f32
constexpr size_t OFF_MISC = OFF_SABG + (size_t)BN*16*64*sizeof(float);  // [b][16][16] f32

// ---- WMMA fragment helpers (CDNA5 16x16x32 f16 layouts, wave32) ----
// A (16x32, 16-bit): lanes 0-15 hold M=lane, K = {k0..k0+7} in v0-3 and
// {k0+16..k0+23} in v4-7; lanes 16-31 hold K offset by +8.
__device__ inline v16h frag_a(const h16* base, int stride, int k0, int lane) {
  int row = lane & 15, half = (lane >> 4) & 1;
  const h16* p0 = base + (size_t)row * stride + k0 + half * 8;
  v8h lo = *(const v8h*)p0;
  v8h hi = *(const v8h*)(p0 + 16);
  v16h a;
#pragma unroll
  for (int i = 0; i < 8; ++i) { a[i] = lo[i]; a[8 + i] = hi[i]; }
  return a;
}
// B (32x16, 16-bit): lanes 0-15: N=lane, K = {k0..k0+7} then {k0+16..k0+23};
// lanes 16-31: N=lane-16, K offset by +8.  LDS/global layout: [col][k].
__device__ inline v16h frag_b(const h16* base, int stride, int k0, int lane) {
  int col = lane & 15, half = (lane >> 4) & 1;
  const h16* p = base + (size_t)col * stride + k0 + half * 8;
  v8h lo = *(const v8h*)p;
  v8h hi = *(const v8h*)(p + 16);
  v16h bfr;
#pragma unroll
  for (int i = 0; i < 8; ++i) { bfr[i] = lo[i]; bfr[8 + i] = hi[i]; }
  return bfr;
}
__device__ inline v8f wmma16(v16h a, v16h b, v8f c) {
  return __builtin_amdgcn_wmma_f32_16x16x32_f16(false, a, false, b, (short)0, c,
                                                false, false);
}

// ---------------- kernel A: fold weights, init slots/positions ----------------
__global__ void kA_setup(const float* Wg, const float* bg, const float* Wk_fg,
                         const float* Wv_fg, const float* Wk_bg, const float* Wv_bg,
                         const float* Wmlp_fg, const float* Wmlp_bg,
                         const float* Wq_init, const float* bq_init,
                         const float* slots_init_bg, const float* fg_pos_init,
                         char* ws) {
  int tid = threadIdx.x;
  h16*   ah   = (h16*)(ws + OFF_AH);
  float* wgmf = (float*)(ws + OFF_WGMF);
  float* wgmb = (float*)(ws + OFF_WGMB);
  float* bgmf = (float*)(ws + OFF_BGMF);
  float* bgmb = (float*)(ws + OFF_BGMB);
  float* sb   = (float*)(ws + OFF_SB);
  float* ppos = (float*)(ws + OFF_PPOS);

  for (int idx = tid; idx < 4 * 64 * 64; idx += blockDim.x) {
    int mat = idx >> 12, o = (idx >> 6) & 63, i = idx & 63;
    const float* wm = (mat < 2) ? Wmlp_fg : Wmlp_bg;
    const float* wx = (mat == 0) ? Wk_fg : (mat == 1) ? Wv_fg : (mat == 2) ? Wk_bg : Wv_bg;
    float acc = 0.f;
    for (int m = 0; m < 64; ++m) acc += wm[o * 64 + m] * wx[m * 64 + i];
    ah[idx] = (h16)acc;
  }
  for (int idx = tid; idx < 2 * 64 * 4; idx += blockDim.x) {
    int grp = idx >> 8, o = (idx >> 2) & 63, j = idx & 3;
    const float* wm = grp ? Wmlp_bg : Wmlp_fg;
    float acc = 0.f;
    for (int m = 0; m < 64; ++m) acc += wm[o * 64 + m] * Wg[m * 4 + j];
    (grp ? wgmb : wgmf)[o * 4 + j] = acc;
  }
  for (int idx = tid; idx < 2 * 64; idx += blockDim.x) {
    int grp = idx >> 6, o = idx & 63;
    const float* wm = grp ? Wmlp_bg : Wmlp_fg;
    float acc = 0.f;
    for (int m = 0; m < 64; ++m) acc += wm[o * 64 + m] * bg[m];
    (grp ? bgmb : bgmf)[o] = acc;
  }
  for (int idx = tid; idx < BN * 16 * 64; idx += blockDim.x) {
    int b = idx >> 10, s = (idx >> 6) & 15, o = idx & 63;
    float v = 0.f;
    if (s == 0) v = slots_init_bg[o];
    else if (s <= KFG) {
      float px = fg_pos_init[(b * KFG + s - 1) * 2 + 0];
      float py = fg_pos_init[(b * KFG + s - 1) * 2 + 1];
      v = px * Wq_init[o * 2 + 0] + py * Wq_init[o * 2 + 1] + bq_init[o];
    }
    sb[idx] = v;
  }
  for (int idx = tid; idx < BN * KFG * 2; idx += blockDim.x) ppos[idx] = fg_pos_init[idx];
}

// ---- kernel B: LN(feat), 4 WMMA GEMMs -> KK/KBG ([n][d]) and VV/VBG ([d][n]),
//      plus misc operand rows (grid, LN(color), Wpos) -------------------------
__global__ void kB_pre(const float* feat, const float* feat_color,
                       const float* g_feat, const float* b_feat,
                       const float* g_color, const float* b_color,
                       const float* Wpos, char* ws) {
  __shared__ h16 X[16 * 64];
  int lane = threadIdx.x;
  int b = blockIdx.x >> 8;
  int nbase = (blockIdx.x & 255) * 16;
  const h16*   ah   = (const h16*)(ws + OFF_AH);
  const float* wgmf = (const float*)(ws + OFF_WGMF);
  const float* wgmb = (const float*)(ws + OFF_WGMB);
  const float* bgmf = (const float*)(ws + OFF_BGMF);
  const float* bgmb = (const float*)(ws + OFF_BGMB);
  h16* kkh  = (h16*)(ws + OFF_KKH);
  h16* kbgh = (h16*)(ws + OFF_KBGH);
  h16* vvt  = (h16*)(ws + OFF_VVT);
  h16* vbgt = (h16*)(ws + OFF_VBGT);
  h16* fct  = (h16*)(ws + OFF_FCT);

  if (lane < 16) {
    int n = nbase + lane;
    const float* fr = feat + ((size_t)b * NPIX + n) * 64;
    float s = 0.f, sq = 0.f;
    for (int d = 0; d < 64; ++d) { float x = fr[d]; s += x; sq += x * x; }
    float m = s * (1.f / 64.f);
    float rstd = rsqrtf(sq * (1.f / 64.f) - m * m + LNEPS);
    for (int d = 0; d < 64; ++d)
      X[lane * 64 + d] = (h16)((fr[d] - m) * rstd * g_feat[d] + b_feat[d]);
    // misc operand rows
    const float* cr = feat_color + ((size_t)b * NPIX + n) * 8;
    float cv[8], cs = 0.f, csq = 0.f;
    for (int c = 0; c < 8; ++c) { cv[c] = cr[c]; cs += cv[c]; csq += cv[c] * cv[c]; }
    float cm = cs * 0.125f;
    float crs = rsqrtf(csq * 0.125f - cm * cm + LNEPS);
    int hh = n >> 6, ww = n & 63;
    float gx = -1.f + (2.f / 63.f) * ww, gy = -1.f + (2.f / 63.f) * hh;
    h16* frow = fct + (size_t)b * 16 * NPIX;
    frow[0 * NPIX + n] = (h16)gx;
    frow[1 * NPIX + n] = (h16)gy;
    for (int c = 0; c < 8; ++c)
      frow[(2 + c) * NPIX + n] = (h16)((cv[c] - cm) * crs * g_color[c] + b_color[c]);
    frow[10 * NPIX + n] = (h16)Wpos[n];
    frow[11 * NPIX + n] = (h16)Wpos[NPIX + n];
    for (int r = 12; r < 16; ++r) frow[r * NPIX + n] = (h16)0.f;
  }
  __syncthreads();

  v16h a0 = frag_a(X, 64, 0, lane);
  v16h a1 = frag_a(X, 64, 32, lane);
  int col = lane & 15, mb = (lane >> 4) * 8;
#pragma unroll
  for (int mat = 0; mat < 4; ++mat) {
    const h16*   A    = ah + mat * 64 * 64;
    const float* wgm  = (mat < 2) ? wgmf : wgmb;
    const float* bgmv = (mat < 2) ? bgmf : bgmb;
#pragma unroll
    for (int t = 0; t < 4; ++t) {
      v8f d = {};
      d = wmma16(a0, frag_b(A + t * 16 * 64, 64, 0, lane), d);
      d = wmma16(a1, frag_b(A + t * 16 * 64, 64, 32, lane), d);
      int o = t * 16 + col;
      float w0 = wgm[o * 4 + 0], w1 = wgm[o * 4 + 1];
      float w2 = wgm[o * 4 + 2], w3 = wgm[o * 4 + 3];
      float bb = bgmv[o];
#pragma unroll
      for (int r = 0; r < 8; ++r) {
        int n = nbase + mb + r;
        int hh = n >> 6, ww = n & 63;
        float gx = -1.f + (2.f / 63.f) * ww, gy = -1.f + (2.f / 63.f) * hh;
        h16 hv = (h16)(d[r] + gx * (w0 - w2) + gy * (w1 - w3) + bb);
        if (mat == 0)      kkh[((size_t)b * NPIX + n) * 64 + o] = hv;
        else if (mat == 1) vvt[((size_t)b * 64 + o) * NPIX + n] = hv;
        else if (mat == 2) kbgh[((size_t)b * NPIX + n) * 64 + o] = hv;
        else               vbgt[((size_t)b * 64 + o) * NPIX + n] = hv;
      }
    }
  }
}

// ---- kernel C (per iter): q = LN(slots)@W.T, Pm, cbias, zero accumulators ----
__global__ void kC_q(const float* g_q, const float* b_q, const float* Wq,
                     const float* g_qbg, const float* b_qbg, const float* Wq_bg,
                     char* ws) {
  __shared__ float smV[64], smX[64];
  int tid = threadIdx.x, b = blockIdx.x;
  float* sb   = (float*)(ws + OFF_SB);
  h16*   qh   = (h16*)(ws + OFF_QH);
  float* pm   = (float*)(ws + OFF_PM);
  float* cb   = (float*)(ws + OFF_CB);
  float* ppos = (float*)(ws + OFF_PPOS);
  const float* wgmf = (const float*)(ws + OFF_WGMF);
  float* ssum = (float*)(ws + OFF_SSUM);
  float* safg = (float*)(ws + OFF_SAFG);
  float* sabg = (float*)(ws + OFF_SABG);
  float* misc = (float*)(ws + OFF_MISC);

  for (int i = tid; i < 16; i += 64) { ssum[b * 16 + i] = 0.f; cb[b * 16 + i] = 0.f; }
  for (int i = tid; i < 16 * 64; i += 64) { safg[b * 1024 + i] = 0.f; sabg[b * 1024 + i] = 0.f; }
  for (int i = tid; i < 256; i += 64) misc[b * 256 + i] = 0.f;
  for (int k = 0; k < KFG; ++k) {
    float px = ppos[(b * KFG + k) * 2 + 0], py = ppos[(b * KFG + k) * 2 + 1];
    float w0 = wgmf[tid * 4 + 0], w1 = wgmf[tid * 4 + 1];
    float w2 = wgmf[tid * 4 + 2], w3 = wgmf[tid * 4 + 3];
    pm[((size_t)b * KFG + k) * 64 + tid] = px * (w0 - w2) + py * (w1 - w3);
  }
  for (int s = NSL; s < 16; ++s) qh[((size_t)b * 16 + s) * 64 + tid] = (h16)0.f;

  for (int s = 0; s < NSL; ++s) {
    const float* gv = s ? g_q : g_qbg;
    const float* bv = s ? b_q : b_qbg;
    const float* W  = s ? Wq  : Wq_bg;
    float x = sb[((size_t)b * 16 + s) * 64 + tid];
    __syncthreads();
    smV[tid] = x;
    __syncthreads();
    float su = 0.f, sq = 0.f;
    for (int d = 0; d < 64; ++d) { float t = smV[d]; su += t; sq += t * t; }
    float m = su / 64.f;
    float rstd = rsqrtf(sq / 64.f - m * m + LNEPS);
    float xn = (x - m) * rstd * gv[tid] + bv[tid];
    __syncthreads();
    smX[tid] = xn;
    __syncthreads();
    float q = 0.f;
    for (int d = 0; d < 64; ++d) q += smX[d] * W[tid * 64 + d];
    qh[((size_t)b * 16 + s) * 64 + tid] = (h16)q;
    if (s >= 1) {
      __syncthreads();
      smV[tid] = q * pm[((size_t)b * KFG + (s - 1)) * 64 + tid];
      __syncthreads();
      if (tid == 0) {
        float c = 0.f;
        for (int d = 0; d < 64; ++d) c += smV[d];
        cb[b * 16 + s] = c;
      }
    }
  }
}

// ---- kernel D (per iter): WMMA logits + softmax over 9 slots + attn sums ----
__global__ void kD_attn(char* ws, float* dout, int last) {
  __shared__ float smD[2 * 256];
  __shared__ float smR[16 * NSL];
  int lane = threadIdx.x;
  int b = blockIdx.x >> 8;
  int nbase = (blockIdx.x & 255) * 16;
  const h16* qh    = (const h16*)(ws + OFF_QH) + (size_t)b * 16 * 64;
  const h16* kkh   = (const h16*)(ws + OFF_KKH) + (size_t)b * NPIX * 64;
  const h16* kbgh  = (const h16*)(ws + OFF_KBGH) + (size_t)b * NPIX * 64;
  const float* cb  = (const float*)(ws + OFF_CB) + b * 16;
  h16*   attnw = (h16*)(ws + OFF_ATTN) + (size_t)b * 16 * NPIX;
  float* ssum  = (float*)(ws + OFF_SSUM) + b * 16;

  v16h a0 = frag_a(qh, 64, 0, lane);
  v16h a1 = frag_a(qh, 64, 32, lane);
  v8f dfg = {}, dbg = {};
  dfg = wmma16(a0, frag_b(kkh + (size_t)nbase * 64, 64, 0, lane), dfg);
  dfg = wmma16(a1, frag_b(kkh + (size_t)nbase * 64, 64, 32, lane), dfg);
  dbg = wmma16(a0, frag_b(kbgh + (size_t)nbase * 64, 64, 0, lane), dbg);
  dbg = wmma16(a1, frag_b(kbgh + (size_t)nbase * 64, 64, 32, lane), dbg);
  int col = lane & 15, mb = (lane >> 4) * 8;
#pragma unroll
  for (int r = 0; r < 8; ++r) {
    smD[(mb + r) * 16 + col] = dfg[r];
    smD[256 + (mb + r) * 16 + col] = dbg[r];
  }
  __syncthreads();
  if (lane < 16) {
    int n = nbase + lane;
    float lg[NSL];
    lg[0] = smD[256 + lane] * SCALEF;
    for (int s = 1; s < NSL; ++s) lg[s] = (smD[s * 16 + lane] - cb[s]) * SCALEF;
    float mx = lg[0];
    for (int s = 1; s < NSL; ++s) mx = fmaxf(mx, lg[s]);
    float se = 0.f;
    for (int s = 0; s < NSL; ++s) { lg[s] = expf(lg[s] - mx); se += lg[s]; }
    float inv = 1.f / se;
    for (int s = 0; s < NSL; ++s) {
      float at = lg[s] * inv + EPSF;
      attnw[(size_t)s * NPIX + n] = (h16)at;
      if (last) dout[OUT_ATTN + ((size_t)(b * NSL + s)) * NPIX + n] = at;
      smR[lane * NSL + s] = at;
    }
    for (int s = NSL; s < 16; ++s) attnw[(size_t)s * NPIX + n] = (h16)0.f;
  }
  __syncthreads();
  if (lane < NSL) {
    float t = 0.f;
    for (int c = 0; c < 16; ++c) t += smR[c * NSL + lane];
    atomicAdd(&ssum[lane], t);
  }
}

// ---- kernel E (per iter): WMMA K-reductions attn@VV, attn@VBG, attn@misc ----
__global__ void kE_acc(char* ws) {
  int lane = threadIdx.x;
  int b = blockIdx.x >> 4;
  int kbase = (blockIdx.x & 15) * 256;
  const h16* attnw = (const h16*)(ws + OFF_ATTN) + (size_t)b * 16 * NPIX;
  const h16* vvt   = (const h16*)(ws + OFF_VVT) + (size_t)b * 64 * NPIX;
  const h16* vbgt  = (const h16*)(ws + OFF_VBGT) + (size_t)b * 64 * NPIX;
  const h16* fct   = (const h16*)(ws + OFF_FCT) + (size_t)b * 16 * NPIX;
  float* safg = (float*)(ws + OFF_SAFG) + b * 1024;
  float* sabg = (float*)(ws + OFF_SABG) + b * 1024;
  float* misc = (float*)(ws + OFF_MISC) + b * 256;

  v8f dfg[4] = {v8f{}, v8f{}, v8f{}, v8f{}};
  v8f dbg[4] = {v8f{}, v8f{}, v8f{}, v8f{}};
  v8f dm = {};
  for (int ks = 0; ks < 8; ++ks) {
    int k0 = kbase + ks * 32;
    v16h a = frag_a(attnw, NPIX, k0, lane);
    dm = wmma16(a, frag_b(fct, NPIX, k0, lane), dm);
#pragma unroll
    for (int t = 0; t < 4; ++t) {
      dfg[t] = wmma16(a, frag_b(vvt + (size_t)t * 16 * NPIX, NPIX, k0, lane), dfg[t]);
      dbg[t] = wmma16(a, frag_b(vbgt + (size_t)t * 16 * NPIX, NPIX, k0, lane), dbg[t]);
    }
  }
  int col = lane & 15, mb = (lane >> 4) * 8;
#pragma unroll
  for (int r = 0; r < 8; ++r) {
    int m = mb + r;
    atomicAdd(&misc[m * 16 + col], dm[r]);
#pragma unroll
    for (int t = 0; t < 4; ++t) {
      atomicAdd(&safg[m * 64 + t * 16 + col], dfg[t][r]);
      atomicAdd(&sabg[m * 64 + t * 16 + col], dbg[t][r]);
    }
  }
}

// ---- kernel F (per iter): position + slot updates, final output assembly ----
__global__ void kF_fin(const float* bpos, char* ws, float* dout, int last) {
  int tid = threadIdx.x, b = blockIdx.x;
  float* sb   = (float*)(ws + OFF_SB) + (size_t)b * 16 * 64;
  float* ppos = (float*)(ws + OFF_PPOS) + b * KFG * 2;
  const float* ssum = (const float*)(ws + OFF_SSUM) + b * 16;
  const float* safg = (const float*)(ws + OFF_SAFG) + b * 1024;
  const float* sabg = (const float*)(ws + OFF_SABG) + b * 1024;
  const float* misc = (const float*)(ws + OFF_MISC) + b * 256;
  const float* pm   = (const float*)(ws + OFF_PM) + (size_t)b * KFG * 64;

  if (tid < 16) {
    int k = tid >> 1, j = tid & 1;
    float S = ssum[k + 1];
    float np = (misc[(k + 1) * 16 + j] / S) * (1.f - MOMF) + ppos[k * 2 + j] * MOMF;
    if (last) {
      np += tanhf(misc[(k + 1) * 16 + 10 + j] / S + bpos[j]) * 0.2f;
      np = fminf(1.f, fmaxf(-1.f, np));
      dout[OUT_POS + (b * KFG + k) * 2 + j] = np;
    }
    ppos[k * 2 + j] = np;
  }
  if (!last) {
    float Sbg = ssum[0];
    sb[tid] += sabg[tid] / Sbg;
    for (int s = 1; s < NSL; ++s)
      sb[s * 64 + tid] += safg[s * 64 + tid] / ssum[s] - pm[(s - 1) * 64 + tid];
  } else {
    float Sbg = ssum[0];
    dout[((size_t)b * NSL) * 72 + tid] = sb[tid];
    if (tid < 8) dout[((size_t)b * NSL) * 72 + 64 + tid] = misc[2 + tid] / Sbg;
    for (int s = 1; s < NSL; ++s) {
      dout[((size_t)b * NSL + s) * 72 + tid] = sb[s * 64 + tid];
      if (tid < 8)
        dout[((size_t)b * NSL + s) * 72 + 64 + tid] = misc[s * 16 + 2 + tid] / ssum[s];
    }
  }
}

extern "C" void kernel_launch(void* const* d_in, const int* in_sizes, int n_in,
                              void* d_out, int out_size, void* d_ws, size_t ws_size,
                              hipStream_t stream) {
  (void)in_sizes; (void)n_in; (void)out_size; (void)ws_size;
  const float* feat          = (const float*)d_in[0];
  const float* feat_color    = (const float*)d_in[1];
  const float* fg_pos_init   = (const float*)d_in[2];
  const float* Wg            = (const float*)d_in[3];
  const float* bg            = (const float*)d_in[4];
  const float* Wk_fg         = (const float*)d_in[5];
  const float* Wv_fg         = (const float*)d_in[6];
  const float* Wk_bg         = (const float*)d_in[7];
  const float* Wv_bg         = (const float*)d_in[8];
  const float* Wmlp_fg       = (const float*)d_in[9];
  const float* Wmlp_bg       = (const float*)d_in[10];
  const float* Wq_init       = (const float*)d_in[11];
  const float* bq_init       = (const float*)d_in[12];
  const float* slots_init_bg = (const float*)d_in[13];
  const float* g_q           = (const float*)d_in[14];
  const float* b_q           = (const float*)d_in[15];
  const float* Wq            = (const float*)d_in[16];
  const float* g_qbg         = (const float*)d_in[17];
  const float* b_qbg         = (const float*)d_in[18];
  const float* Wq_bg         = (const float*)d_in[19];
  const float* g_feat        = (const float*)d_in[20];
  const float* b_feat        = (const float*)d_in[21];
  const float* g_color       = (const float*)d_in[22];
  const float* b_color       = (const float*)d_in[23];
  const float* Wpos          = (const float*)d_in[24];
  const float* bpos          = (const float*)d_in[25];
  char*  ws  = (char*)d_ws;
  float* out = (float*)d_out;

  kA_setup<<<1, 256, 0, stream>>>(Wg, bg, Wk_fg, Wv_fg, Wk_bg, Wv_bg, Wmlp_fg,
                                  Wmlp_bg, Wq_init, bq_init, slots_init_bg,
                                  fg_pos_init, ws);
  kB_pre<<<BN * 256, 32, 0, stream>>>(feat, feat_color, g_feat, b_feat, g_color,
                                      b_color, Wpos, ws);
  for (int it = 0; it < NITER; ++it) {
    int last = (it == NITER - 1);
    kC_q<<<BN, 64, 0, stream>>>(g_q, b_q, Wq, g_qbg, b_qbg, Wq_bg, ws);
    kD_attn<<<BN * 256, 32, 0, stream>>>(ws, out, last);
    kE_acc<<<BN * 16, 32, 0, stream>>>(ws);
    kF_fin<<<BN, 64, 0, stream>>>(bpos, ws, out, last);
  }
}
